// Quantizer_43215960932579
// MI455X (gfx1250) — matrix operand order repaired and checked by reading, not compiled, not run on hardware
//
#include <hip/hip_runtime.h>

typedef __attribute__((ext_vector_type(16))) _Float16     v16h;
typedef __attribute__((ext_vector_type(8)))  float        v8f;
typedef __attribute__((ext_vector_type(4)))  unsigned int v4u;
typedef __attribute__((ext_vector_type(8)))  int          v8i_;
typedef __attribute__((ext_vector_type(4)))  int          v4i_;

#define KCODES 1024
#define DIM    64
#define NROWS  32768          // 32*32*32
#define MTOT   2097152        // NROWS*DIM
#define BETA   0.25f
#define DECAY  0.99f
#define EPSV   1e-5f

// ---- workspace layout (float offsets into d_ws) ----
#define WS_COUNTS 0           // 1024
#define WS_DW     1024        // 65536
#define WS_VX     66560       // 64  (sum of xr rows)
#define WS_VE     66624       // 64  (sum of gathered new-embedding rows)
#define WS_SSE    66688
#define WS_S2     66689
#define WS_S1     66690
#define WS_NSUM   66691
#define WS_ENT    66692
#define WS_ZEND   66816       // zeroed region [0, WS_ZEND)
#define WS_ENORM  66816       // 1024
#define WS_IDX    67840       // 32768 ints
#define WS_EH     100608      // codebook f16: 65536 halfs (32768 float slots)
#define WS_XRH    133376      // xr f16: 2097152 halfs (1048576 float slots)
#define WS_XR     1181952     // xr f32: 2097152 floats

// ---- output layout (float offsets into d_out, tuple order) ----
#define OUT_Q     0
#define OUT_CB    2097152
#define OUT_CM    2097153
#define OUT_DV    2097154
#define OUT_PP    2097155
#define OUT_IDX   2097156
#define OUT_CS    2129924
#define OUT_EMA   2130948
#define OUT_EMB   2196484

// ---------------- TDM helpers ----------------
// Stage 16KB (128 codes x 128B, viewed as 4096 dwords) of the f16 codebook
// into LDS via the Tensor Data Mover. D# per CDNA5 ISA:
// group0: count=1, lds_addr, global_addr[56:0], type=2
// group1: data_size=2 (4B), tensor_dim0=32768, tile_dim0=4096, 1-D tile
__device__ __forceinline__ void tdm_stage_chunk(unsigned long long gaddr,
                                                unsigned lds_addr) {
    v4u g0;
    g0.x = 1u;                                            // count=1 (valid)
    g0.y = lds_addr;                                      // LDS byte address
    g0.z = (unsigned)gaddr;                               // global_addr[31:0]
    g0.w = (unsigned)((gaddr >> 32) & 0x01FFFFFFu) | (2u << 30);  // [56:32] | type=2
    v8i_ g1;
    g1[0] = 0x20000;                                      // data_size=2 (4 bytes)
    g1[1] = (int)0x80000000u;                             // tensor_dim0(=32768) lo16 << 16
    g1[2] = 0x10000;                                      // tensor_dim0 hi16=0 | tensor_dim1=1
    g1[3] = 0x10000000;                                   // tile_dim0 = 4096 << 16
    g1[4] = 0;                                            // tile_dim1=0, tile_dim2=0 (1-D)
    g1[5] = 32768;                                        // tensor_dim0_stride lo32
    g1[6] = 0;                                            // stride hi | dim1_stride lo
    g1[7] = 0;
#if __has_builtin(__builtin_amdgcn_tensor_load_to_lds)
    // 6-arg form (clang-23 / therock-10.0): (v4u, v8i, v4i, v4i, v8i, i32 cpol)
    v4i_ g2 = {1, 1, 0, 0};                               // tensor_dim2=1, dim3=1 (unused)
    v4i_ g3 = {0, 1 << 16, 0, 0};                         // tensor_dim4=1 (unused)
    v8i_ g4 = {0, 0, 0, 0, 0, 0, 0, 0};                   // zero-filled extra group
    __builtin_amdgcn_tensor_load_to_lds(g0, g1, g2, g3, g4, 0);
#else
    asm volatile("tensor_load_to_lds %0, %1" :: "s"(g0), "s"(g1) : "memory");
#endif
}

__device__ __forceinline__ void wait_tensor0() {
#if __has_builtin(__builtin_amdgcn_s_wait_tensorcnt)
    __builtin_amdgcn_s_wait_tensorcnt(0);
#else
    asm volatile("s_wait_tensorcnt 0x0" ::: "memory");
#endif
}

__global__ __launch_bounds__(256) void k_zero(float* ws) {
    int i = blockIdx.x * 256 + threadIdx.x;
    if (i < WS_ZEND) ws[i] = 0.f;
}

// codebook -> f16 (row-major [K,64]) + ||e_k||^2
__global__ __launch_bounds__(256) void k_prep_codebook(const float* __restrict__ E, float* ws) {
    int k = blockIdx.x * 256 + threadIdx.x;
    if (k >= KCODES) return;
    _Float16* eh = (_Float16*)(ws + WS_EH);
    float s = 0.f;
    #pragma unroll 8
    for (int d = 0; d < DIM; ++d) {
        float v = E[k * DIM + d];
        s += v * v;
        eh[k * DIM + d] = (_Float16)v;
    }
    ws[WS_ENORM + k] = s;
}

// xr = transpose(x)(N,D) @ R ; emit f32 + f16, accumulate S2 and vx
__global__ __launch_bounds__(256) void k_rotate(const float* __restrict__ x,
                                                const float* __restrict__ R,
                                                float* ws) {
    __shared__ float xs[32][64];
    __shared__ float Rl[64][64];
    __shared__ float colsum[64];
    __shared__ float s2loc;
    int t = threadIdx.x, blk = blockIdx.x;
    for (int i = t; i < 2048; i += 256) {
        int r = i >> 6, c = i & 63;
        int n = blk * 32 + r;
        int b = n >> 10, hw = n & 1023;
        xs[r][c] = x[b * 65536 + c * 1024 + hw];
    }
    for (int i = t; i < 4096; i += 256) Rl[i >> 6][i & 63] = R[i];
    if (t < 64) colsum[t] = 0.f;
    if (t == 0) s2loc = 0.f;
    __syncthreads();

    int r = t >> 3;               // 8 threads per row
    int j0 = (t & 7) * 8;         // 8 output columns each
    int n = blk * 32 + r;
    float acc[8];
    #pragma unroll
    for (int jj = 0; jj < 8; ++jj) acc[jj] = 0.f;
    for (int c = 0; c < 64; ++c) {
        float xv = xs[r][c];
        #pragma unroll
        for (int jj = 0; jj < 8; ++jj) acc[jj] = fmaf(xv, Rl[c][j0 + jj], acc[jj]);
    }
    float ss = 0.f;
    _Float16* xrh = (_Float16*)(ws + WS_XRH);
    #pragma unroll
    for (int jj = 0; jj < 8; ++jj) {
        float v = acc[jj];
        ss += v * v;
        ws[WS_XR + n * 64 + j0 + jj] = v;
        xrh[n * 64 + j0 + jj] = (_Float16)v;
        atomicAdd(&colsum[j0 + jj], v);
    }
    atomicAdd(&s2loc, ss);
    __syncthreads();
    if (t < 64) atomicAdd(&ws[WS_VX + t], colsum[t]);
    if (t == 0) atomicAdd(&ws[WS_S2], s2loc);
}

// WMMA distance GEMM + argmin with TDM-staged, double-buffered codebook.
// 8 waves/block share the LDS codebook chunk; wave 0 drives the TDM pipeline.
__global__ __launch_bounds__(256) void k_argmin(float* ws, float* __restrict__ out) {
    __shared__ uint4 cb[2][1024];                // 2 x 16KB: 128 codes x 128B each

    int lane = threadIdx.x & 31;
    int wave = threadIdx.x >> 5;
    int rowTile = blockIdx.x * 8 + wave;         // 2048 tiles of 16 rows
    int row0 = rowTile * 16;
    int halfSel = lane >> 4;                     // K-half select
    int laneM = lane & 15;                       // A: row M; B/D: column N
    int row = row0 + laneM;

    const uint4* A4 = (const uint4*)(ws + WS_XRH);   // 8 uint4 per 64-half row
    const float* enorm = ws + WS_ENORM;
    unsigned long long ebase =
        (unsigned long long)(uintptr_t)((const char*)ws + (size_t)WS_EH * 4);

    // A fragment, 16x32 f16, two k-steps (K=0..31, 32..63):
    union { v16h h; uint4 u[2]; } a0, a1;
    a0.u[0] = A4[row * 8 + 0 + halfSel];
    a0.u[1] = A4[row * 8 + 0 + halfSel + 2];
    a1.u[0] = A4[row * 8 + 4 + halfSel];
    a1.u[1] = A4[row * 8 + 4 + halfSel + 2];

    float best[8];
    int   bidx[8];
    #pragma unroll
    for (int v = 0; v < 8; ++v) { best[v] = 3.4e38f; bidx[v] = 0; }

    bool issuer = (wave == 0);
    if (issuer) {
        tdm_stage_chunk(ebase, (unsigned)(uintptr_t)&cb[0][0]);
        wait_tensor0();
    }
    __syncthreads();

    for (int c = 0; c < 8; ++c) {                // 8 chunks of 128 codes
        if (issuer && c + 1 < 8)
            tdm_stage_chunk(ebase + (unsigned long long)(c + 1) * 16384ull,
                            (unsigned)(uintptr_t)&cb[(c + 1) & 1][0]);
        if (c + 1 < 8)
            __builtin_prefetch(&enorm[(c + 1) * 128 + laneM], 0, 1);

        const uint4* buf = &cb[c & 1][0];
        #pragma unroll 2
        for (int t = 0; t < 8; ++t) {            // 8 code tiles per chunk
            int lc = t * 16 + laneM;             // local code in chunk
            int code = c * 128 + lc;
            // B fragment 32x16 f16 from LDS: lane=column,
            // lanes<16 K=0..7,16..23, lanes>=16 K=8..15,24..31 (+32 per k-step)
            union { v16h h; uint4 u[2]; } b0, b1;
            int base = lc * 8 + halfSel;
            b0.u[0] = buf[base + 0];
            b0.u[1] = buf[base + 2];
            b1.u[0] = buf[base + 4];
            b1.u[1] = buf[base + 6];

            v8f cacc = {0.f, 0.f, 0.f, 0.f, 0.f, 0.f, 0.f, 0.f};
            cacc = __builtin_amdgcn_wmma_f32_16x16x32_f16(false, a0.h, false, b0.h,
                                                          (short)0, cacc, false, false);
            cacc = __builtin_amdgcn_wmma_f32_16x16x32_f16(false, a1.h, false, b1.h,
                                                          (short)0, cacc, false, false);
            float en = enorm[code];
            #pragma unroll
            for (int v = 0; v < 8; ++v) {        // element: row v+8*halfSel, col laneM
                float d = en - 2.f * cacc[v];
                if (d < best[v]) { best[v] = d; bidx[v] = code; }
            }
        }
        if (c + 1 < 8) {
            if (issuer) wait_tensor0();          // next chunk landed
            __syncthreads();                     // everyone done with buf, DMA visible
        }
    }

    // reduce across the 16 column-lanes of each half (xor masks keep bit4 fixed)
    #pragma unroll
    for (int v = 0; v < 8; ++v) {
        for (int m = 1; m < 16; m <<= 1) {
            float ob = __shfl_xor(best[v], m, 32);
            int   oi = __shfl_xor(bidx[v], m, 32);
            if (ob < best[v] || (ob == best[v] && oi < bidx[v])) { best[v] = ob; bidx[v] = oi; }
        }
    }
    if (laneM == 0) {                            // lane 0 -> rows 0..7, lane 16 -> rows 8..15
        int* idxbuf = (int*)ws + WS_IDX;
        #pragma unroll
        for (int v = 0; v < 8; ++v) {
            int rn = row0 + halfSel * 8 + v;
            idxbuf[rn] = bidx[v];
            out[OUT_IDX + rn] = (float)bidx[v];
        }
    }
}

// quant_st gather (old codebook), SSE for both losses, dw/counts scatter-add
__global__ __launch_bounds__(256) void k_scatter(const float* __restrict__ x,
                                                 const float* __restrict__ E,
                                                 float* ws, float* __restrict__ out) {
    __shared__ float sseloc;
    if (threadIdx.x == 0) sseloc = 0.f;
    __syncthreads();
    int j = blockIdx.x * 256 + threadIdx.x;      // flat over N*D; matches x flat order
    int n = j >> 6, d = j & 63;
    int k = ((const int*)ws + WS_IDX)[n];
    float q = E[k * 64 + d];
    out[OUT_Q + j] = q;                          // quant_st == quant_out (straight-through)
    float diff = q - x[j];
    atomicAdd(&sseloc, diff * diff);
    atomicAdd(&ws[WS_DW + k * 64 + d], ws[WS_XR + n * 64 + d]);
    if (d == 0) atomicAdd(&ws[WS_COUNTS + k], 1.0f);
    __syncthreads();
    if (threadIdx.x == 0) atomicAdd(&ws[WS_SSE], sseloc);
}

// cluster-size normalization sum + entropy (single block, K=1024)
__global__ __launch_bounds__(1024) void k_final1(const float* __restrict__ ema_cs, float* ws) {
    __shared__ float nsum, ent;
    int t = threadIdx.x;
    if (t == 0) { nsum = 0.f; ent = 0.f; }
    __syncthreads();
    float cnt  = ws[WS_COUNTS + t];
    float craw = ema_cs[t] * DECAY + (1.f - DECAY) * cnt;
    float p    = cnt / (float)NROWS;
    atomicAdd(&nsum, craw);
    atomicAdd(&ent, p * __logf(p + 1e-10f));
    __syncthreads();
    if (t == 0) { ws[WS_NSUM] = nsum; ws[WS_ENT] = ent; }
}

// EMA update, new embedding, cs output; accumulate S1 and ve for diversity
__global__ __launch_bounds__(256) void k_final2(const float* __restrict__ ema_w,
                                                const float* __restrict__ ema_cs,
                                                float* ws, float* __restrict__ out) {
    __shared__ float s1loc;
    if (threadIdx.x == 0) s1loc = 0.f;
    __syncthreads();
    int e = blockIdx.x * 256 + threadIdx.x;      // over K*D = 65536
    int k = e >> 6, d = e & 63;
    float cnt  = ws[WS_COUNTS + k];
    float craw = ema_cs[k] * DECAY + (1.f - DECAY) * cnt;
    float nsum = ws[WS_NSUM];
    float cs   = (craw + EPSV) / (nsum + KCODES * EPSV) * nsum;
    float nema = ema_w[e] * DECAY + (1.f - DECAY) * ws[WS_DW + e];
    float emb  = nema / cs;
    out[OUT_EMA + e] = nema;
    out[OUT_EMB + e] = emb;
    if (d == 0) out[OUT_CS + k] = cs;
    atomicAdd(&s1loc, cnt * emb * emb);          // S1 = sum_k counts[k]*||Enew_k||^2
    atomicAdd(&ws[WS_VE + d], cnt * emb);        // ve = sum_n Enew[idx_n]
    __syncthreads();
    if (threadIdx.x == 0) atomicAdd(&ws[WS_S1], s1loc);
}

// scalar epilogue: losses, diversity (closed form), perplexity
__global__ __launch_bounds__(64) void k_final3(float* ws, float* __restrict__ out) {
    __shared__ float dotv;
    int t = threadIdx.x;
    if (t == 0) dotv = 0.f;
    __syncthreads();
    atomicAdd(&dotv, ws[WS_VE + t] * ws[WS_VX + t]);
    __syncthreads();
    if (t == 0) {
        float cb = ws[WS_SSE] / (float)MTOT;
        out[OUT_CB] = cb;
        out[OUT_CM] = BETA * cb;
        float Nf = (float)NROWS;
        out[OUT_DV] = ws[WS_S1] / Nf + ws[WS_S2] / Nf - 2.f * dotv / (Nf * Nf);
        out[OUT_PP] = __expf(-ws[WS_ENT]);
    }
}

extern "C" void kernel_launch(void* const* d_in, const int* in_sizes, int n_in,
                              void* d_out, int out_size, void* d_ws, size_t ws_size,
                              hipStream_t stream) {
    const float* x      = (const float*)d_in[0];
    const float* E      = (const float*)d_in[1];
    const float* R      = (const float*)d_in[2];
    const float* ema_w  = (const float*)d_in[3];
    const float* ema_cs = (const float*)d_in[4];
    float* out = (float*)d_out;
    float* ws  = (float*)d_ws;

    k_zero<<<(WS_ZEND + 255) / 256, 256, 0, stream>>>(ws);
    k_prep_codebook<<<4, 256, 0, stream>>>(E, ws);
    k_rotate<<<NROWS / 32, 256, 0, stream>>>(x, R, ws);
    k_argmin<<<NROWS / 16 / 8, 256, 0, stream>>>(ws, out);
    k_scatter<<<MTOT / 256, 256, 0, stream>>>(x, E, ws, out);
    k_final1<<<1, 1024, 0, stream>>>(ema_cs, ws);
    k_final2<<<KCODES * DIM / 256, 256, 0, stream>>>(ema_w, ema_cs, ws, out);
    k_final3<<<1, 64, 0, stream>>>(ws, out);
}
